// SphericalPhasorBlock_37082747633870
// MI455X (gfx1250) — matrix operand order repaired and checked by reading, not compiled, not run on hardware
//
#include <hip/hip_runtime.h>
#include <hip/hip_bf16.h>
#include <math.h>

// ---------------------------------------------------------------------------
// SphericalPhasorBlock for MI455X (gfx1250, wave32, WMMA)
//
// Shapes: B=2, L=2048, D=512, K=64 (P=32 complex pairs). M = B*L = 4096.
// GEMMs run as bf16 WMMA (v_wmma_f32_16x16x32_bf16) with f32 accumulation and
// a register ping-pong software pipeline; all B-tile addressing uses one
// moving base pointer + 24-bit immediate offsets (Kd is a template constant)
// so the inner loop carries only 2 address pairs -> no scratch spills.
// The phasor prefix-sum runs as a 3-phase chunked scan with the 128KB complex
// state held in LDS (WGP has 320KB), LayerNorm fused into the scan epilogue.
// ---------------------------------------------------------------------------

typedef __attribute__((ext_vector_type(16))) __bf16 v16bf;
typedef __attribute__((ext_vector_type(8)))  __bf16 v8bf;
typedef __attribute__((ext_vector_type(8)))  float  v8f;

union ABfrag { v16bf v; v8bf h[2]; };

#define DD   512
#define PP   32
#define LL   2048
#define BB   2
#define MM   (BB * LL)
#define CC   32          // scan chunks
#define LCC  (LL / CC)   // positions per chunk = 64

__device__ __forceinline__ unsigned short f2bf(float f) {
  unsigned u = __float_as_uint(f);
  u = (u + 0x7FFFu + ((u >> 16) & 1u)) >> 16;   // round-to-nearest-even
  return (unsigned short)u;
}

// ---------------- pack / transpose to bf16 ----------------
__global__ void pack_bf16_kernel(const float* __restrict__ a,
                                 unsigned short* __restrict__ o, int n) {
  int i = blockIdx.x * blockDim.x + threadIdx.x;
  if (i < n) o[i] = f2bf(a[i]);
}

// w: Din x Dout (row-major)  ->  wT: Dout x Din (row-major), bf16
__global__ void packT_bf16_kernel(const float* __restrict__ w,
                                  unsigned short* __restrict__ wT,
                                  int Din, int Dout) {
  int i = blockIdx.x * blockDim.x + threadIdx.x;
  if (i >= Din * Dout) return;
  int n = i / Din, k = i - n * Din;
  wT[i] = f2bf(w[k * Dout + n]);
}

// ---------------- WMMA GEMM: C = act(A @ Bt^T + bias) [+ res] -------------
// A : M x KD bf16 row-major; Bt : N x KD bf16 row-major (weight transposed)
// block = 256 threads = 8 waves; wave computes 16 rows x 64 cols (4 tiles)
// Fragment layouts per CDNA5 ISA 7.12.2:
//   A(16x32 bf16): lane<16 -> row=lane, K={0..7,16..23}; lane>=16 -> K={8..15,24..31}
//   B(32x16 bf16): lane<16 -> col=lane, K={0..7,16..23}; lane>=16 -> K={8..15,24..31}
template<int ACT_GELU, int OUT_BF16, int ADD_RES, int KD>
__global__ void gemm_wmma_kernel(const unsigned short* __restrict__ A,
                                 const unsigned short* __restrict__ Bt,
                                 const float* __restrict__ bias,
                                 const float* __restrict__ res,
                                 float* __restrict__ outF,
                                 unsigned short* __restrict__ outB,
                                 int N) {
  const int lane = threadIdx.x & 31;
  const int wv   = threadIdx.x >> 5;
  const int lanecol = lane & 15;
  const int rA   = blockIdx.y * 128 + wv * 16 + lanecol;   // A fragment row
  const int colbase = blockIdx.x * 64;
  const int ahalf = (lane < 16) ? 0 : 8;
  const int bhalf = (lane < 16) ? 0 : 8;

  v8f acc[4] = {};

  // moving per-lane base pointers; everything else is immediate offsets
  const unsigned short* Ap = A + (size_t)rA * KD + ahalf;
  const unsigned short* Bp = Bt + (size_t)(colbase + lanecol) * KD + bhalf;

  ABfrag aA, aB, bA[4], bB[4];
  // k-offset immediates: A halves at ko,ko+16; B tile t at ko + t*16*KD (+16)
  aA.h[0] = *(const v8bf*)(const void*)(Ap);
  aA.h[1] = *(const v8bf*)(const void*)(Ap + 16);
#pragma unroll
  for (int t = 0; t < 4; ++t) {
    bA[t].h[0] = *(const v8bf*)(const void*)(Bp + t * 16 * KD);
    bA[t].h[1] = *(const v8bf*)(const void*)(Bp + t * 16 * KD + 16);
  }

#pragma unroll 1
  for (int k0 = 0; k0 < KD; k0 += 64) {
    __builtin_prefetch(Ap + 128, 0, 3);     // next 256B of streamed A row
    // load k0+32 fragments (overlaps with first 4 WMMAs)
    aB.h[0] = *(const v8bf*)(const void*)(Ap + 32);
    aB.h[1] = *(const v8bf*)(const void*)(Ap + 48);
#pragma unroll
    for (int t = 0; t < 4; ++t) {
      bB[t].h[0] = *(const v8bf*)(const void*)(Bp + 32 + t * 16 * KD);
      bB[t].h[1] = *(const v8bf*)(const void*)(Bp + 32 + t * 16 * KD + 16);
    }
#pragma unroll
    for (int t = 0; t < 4; ++t)
      acc[t] = __builtin_amdgcn_wmma_f32_16x16x32_bf16(
          false, aA.v, false, bA[t].v, (short)0, acc[t], false, false);
    Ap += 64;
    Bp += 64;
    if (k0 + 64 < KD) {
      // load k0+64 fragments (overlaps with second 4 WMMAs)
      aA.h[0] = *(const v8bf*)(const void*)(Ap);
      aA.h[1] = *(const v8bf*)(const void*)(Ap + 16);
#pragma unroll
      for (int t = 0; t < 4; ++t) {
        bA[t].h[0] = *(const v8bf*)(const void*)(Bp + t * 16 * KD);
        bA[t].h[1] = *(const v8bf*)(const void*)(Bp + t * 16 * KD + 16);
      }
    }
#pragma unroll
    for (int t = 0; t < 4; ++t)
      acc[t] = __builtin_amdgcn_wmma_f32_16x16x32_bf16(
          false, aB.v, false, bB[t].v, (short)0, acc[t], false, false);
  }

  // ---- epilogue: bias (+gelu) (+residual), f32 or bf16 store ----
#pragma unroll
  for (int t = 0; t < 4; ++t) {
    int col = colbase + 16 * t + lanecol;
    float bs = bias[col];
#pragma unroll
    for (int v = 0; v < 8; ++v) {
      int row = blockIdx.y * 128 + wv * 16 + v + ((lane < 16) ? 0 : 8);
      float val = acc[t][v] + bs;
      if (ACT_GELU) val = 0.5f * val * (1.0f + erff(val * 0.70710678118654752f));
      if (ADD_RES)  val += res[(size_t)row * N + col];
      if (OUT_BF16) outB[(size_t)row * N + col] = f2bf(val);
      else          outF[(size_t)row * N + col] = val;
    }
  }
}

// ---------------- row-wise L2 normalize (cols = 64), in place -------------
__global__ void l2norm_kernel(float* __restrict__ v, int rows) {
  int lane = threadIdx.x & 31, wv = threadIdx.x >> 5;
  int row = blockIdx.x * 8 + wv;
  if (row >= rows) return;
  float* p = v + (size_t)row * 64;
  float a = p[lane], b = p[lane + 32];
  float s = a * a + b * b;
#pragma unroll
  for (int m = 16; m > 0; m >>= 1) s += __shfl_xor(s, m, 32);
  float sc = 1.0f / fmaxf(sqrtf(s), 1e-12f);
  p[lane] = a * sc;
  p[lane + 32] = b * sc;
}

// ---------------- scan phase 1: per-chunk bound sums ----------------------
// state element layout: [p*D + d] real plane, [P*D + p*D + d] imag plane
__global__ void chunk_sum_kernel(const float* __restrict__ kenc,
                                 const float* __restrict__ Vmat,
                                 float* __restrict__ cstate) {
  extern __shared__ float sm[];
  float* Sr = sm;
  float* Si = sm + PP * DD;
  float* kb = sm + 2 * PP * DD;   // 2*P = 64
  const int b = blockIdx.y, c = blockIdx.x, d = threadIdx.x;
#pragma unroll
  for (int p = 0; p < PP; ++p) { Sr[p * DD + d] = 0.f; Si[p * DD + d] = 0.f; }
  __syncthreads();
  const int l0 = c * LCC;
  for (int l = 0; l < LCC; ++l) {
    int row = b * LL + l0 + l;
    if (threadIdx.x < 2 * PP) kb[threadIdx.x] = kenc[(size_t)row * 64 + threadIdx.x];
    __syncthreads();
    float v = Vmat[(size_t)row * DD + d];
    __builtin_prefetch(Vmat + (size_t)(row + 1) * DD + d, 0, 3);
#pragma unroll
    for (int p = 0; p < PP; ++p) {
      Sr[p * DD + d] += kb[2 * p]     * v;
      Si[p * DD + d] += kb[2 * p + 1] * v;
    }
    __syncthreads();
  }
  float* dst = cstate + ((size_t)(b * CC + c)) * (2 * PP * DD);
#pragma unroll
  for (int p = 0; p < PP; ++p) {
    dst[p * DD + d]            = Sr[p * DD + d];
    dst[PP * DD + p * DD + d]  = Si[p * DD + d];
  }
}

// ---------------- scan phase 2: exclusive prefix over chunk states --------
__global__ void chunk_prefix_kernel(float* __restrict__ cstate) {
  const int elems = 2 * PP * DD;
  int b = blockIdx.y;
  int e = blockIdx.x * blockDim.x + threadIdx.x;
  if (e >= elems) return;
  float run = 0.f;
  float* base = cstate + (size_t)b * CC * elems + e;
  for (int c = 0; c < CC; ++c) {
    float t = base[(size_t)c * elems];
    base[(size_t)c * elems] = run;
    run += t;
  }
}

// ---------------- scan phase 3: in-chunk scan + retrieve + LayerNorm ------
__global__ void scan_ln_kernel(const float* __restrict__ kenc,
                               const float* __restrict__ qenc,
                               const float* __restrict__ Vmat,
                               const float* __restrict__ cstate,
                               const float* __restrict__ ln_g,
                               const float* __restrict__ ln_b,
                               unsigned short* __restrict__ rln) {
  extern __shared__ float sm[];
  float* Sr  = sm;
  float* Si  = sm + PP * DD;
  float* kb  = sm + 2 * PP * DD;          // 64
  float* qb  = kb + 2 * PP;               // 64
  float* red = qb + 2 * PP;               // 32
  const int b = blockIdx.y, c = blockIdx.x, d = threadIdx.x;
  const int lane = threadIdx.x & 31, wv = threadIdx.x >> 5;
  const float* src = cstate + ((size_t)(b * CC + c)) * (2 * PP * DD);
#pragma unroll
  for (int p = 0; p < PP; ++p) {
    Sr[p * DD + d] = src[p * DD + d];
    Si[p * DD + d] = src[PP * DD + p * DD + d];
  }
  __syncthreads();
  const float gg = ln_g[d], gb = ln_b[d];
  const int l0 = c * LCC;
  for (int l = 0; l < LCC; ++l) {
    int row = b * LL + l0 + l;
    if (threadIdx.x < 2 * PP) {
      kb[threadIdx.x] = kenc[(size_t)row * 64 + threadIdx.x];
      qb[threadIdx.x] = qenc[(size_t)row * 64 + threadIdx.x];
    }
    __syncthreads();
    float v = Vmat[(size_t)row * DD + d];
    __builtin_prefetch(Vmat + (size_t)(row + 1) * DD + d, 0, 3);
    float acc = 0.f;
#pragma unroll
    for (int p = 0; p < PP; ++p) {
      float sr = Sr[p * DD + d] + kb[2 * p]     * v;
      float si = Si[p * DD + d] + kb[2 * p + 1] * v;
      Sr[p * DD + d] = sr;
      Si[p * DD + d] = si;
      acc += sr * qb[2 * p] + si * qb[2 * p + 1];
    }
    float r = acc * rsqrtf((float)(l0 + l + 1) * (float)PP);
    // LayerNorm over D=512 (16 waves)
    float s1 = r, s2 = r * r;
#pragma unroll
    for (int m = 16; m > 0; m >>= 1) {
      s1 += __shfl_xor(s1, m, 32);
      s2 += __shfl_xor(s2, m, 32);
    }
    if (lane == 0) { red[wv] = s1; red[16 + wv] = s2; }
    __syncthreads();
    float m1 = 0.f, m2 = 0.f;
#pragma unroll
    for (int w = 0; w < 16; ++w) { m1 += red[w]; m2 += red[16 + w]; }
    float mu = m1 * (1.0f / DD);
    float var = m2 * (1.0f / DD) - mu * mu;
    float y = (r - mu) * rsqrtf(var + 1e-5f) * gg + gb;
    rln[(size_t)row * DD + d] = f2bf(y);
    __syncthreads();   // protect kb/qb/red for next iteration
  }
}

// ---------------------------------------------------------------------------
extern "C" void kernel_launch(void* const* d_in, const int* in_sizes, int n_in,
                              void* d_out, int out_size, void* d_ws, size_t ws_size,
                              hipStream_t stream) {
  (void)in_sizes; (void)n_in; (void)out_size; (void)ws_size;
  const float* x    = (const float*)d_in[0];
  const float* k_w1 = (const float*)d_in[1];
  const float* k_b1 = (const float*)d_in[2];
  const float* k_w2 = (const float*)d_in[3];
  const float* k_b2 = (const float*)d_in[4];
  const float* q_w1 = (const float*)d_in[5];
  const float* q_b1 = (const float*)d_in[6];
  const float* q_w2 = (const float*)d_in[7];
  const float* q_b2 = (const float*)d_in[8];
  const float* v_w  = (const float*)d_in[9];
  const float* v_b  = (const float*)d_in[10];
  const float* ln_g = (const float*)d_in[11];
  const float* ln_b = (const float*)d_in[12];
  const float* o_w  = (const float*)d_in[13];
  const float* o_b  = (const float*)d_in[14];
  float* out = (float*)d_out;

  // ---- workspace carve-up (256B aligned) ----
  char* ws = (char*)d_ws;
  size_t off = 0;
  auto take = [&](size_t bytes) -> char* {
    char* p = ws + off;
    off = (off + bytes + 255) & ~(size_t)255;
    return p;
  };
  unsigned short* xb    = (unsigned short*)take((size_t)MM * DD * 2);
  unsigned short* kw1T  = (unsigned short*)take((size_t)DD * DD * 2);
  unsigned short* qw1T  = (unsigned short*)take((size_t)DD * DD * 2);
  unsigned short* vwT   = (unsigned short*)take((size_t)DD * DD * 2);
  unsigned short* owT   = (unsigned short*)take((size_t)DD * DD * 2);
  unsigned short* kw2T  = (unsigned short*)take((size_t)64 * DD * 2);
  unsigned short* qw2T  = (unsigned short*)take((size_t)64 * DD * 2);
  unsigned short* hk    = (unsigned short*)take((size_t)MM * DD * 2);
  unsigned short* hq    = (unsigned short*)take((size_t)MM * DD * 2);
  float* kenc           = (float*)take((size_t)MM * 64 * 4);
  float* qenc           = (float*)take((size_t)MM * 64 * 4);
  float* Vmat           = (float*)take((size_t)MM * DD * 4);
  float* cstate         = (float*)take((size_t)BB * CC * 2 * PP * DD * 4);
  unsigned short* rln   = (unsigned short*)take((size_t)MM * DD * 2);

  // ---- 0: pack inputs to bf16 ----
  {
    int n = MM * DD;
    pack_bf16_kernel<<<(n + 255) / 256, 256, 0, stream>>>(x, xb, n);
    int nw = DD * DD;
    packT_bf16_kernel<<<(nw + 255) / 256, 256, 0, stream>>>(k_w1, kw1T, DD, DD);
    packT_bf16_kernel<<<(nw + 255) / 256, 256, 0, stream>>>(q_w1, qw1T, DD, DD);
    packT_bf16_kernel<<<(nw + 255) / 256, 256, 0, stream>>>(v_w,  vwT,  DD, DD);
    packT_bf16_kernel<<<(nw + 255) / 256, 256, 0, stream>>>(o_w,  owT,  DD, DD);
    int nw2 = DD * 64;
    packT_bf16_kernel<<<(nw2 + 255) / 256, 256, 0, stream>>>(k_w2, kw2T, DD, 64);
    packT_bf16_kernel<<<(nw2 + 255) / 256, 256, 0, stream>>>(q_w2, qw2T, DD, 64);
  }

  // ---- 1: encoder GEMMs (WMMA bf16) ----
  dim3 gBig(DD / 64, MM / 128);   // (8, 32)
  dim3 gEnc(64 / 64, MM / 128);   // (1, 32)
  gemm_wmma_kernel<1, 1, 0, DD><<<gBig, 256, 0, stream>>>(xb, kw1T, k_b1, nullptr, nullptr, hk, DD);
  gemm_wmma_kernel<1, 1, 0, DD><<<gBig, 256, 0, stream>>>(xb, qw1T, q_b1, nullptr, nullptr, hq, DD);
  gemm_wmma_kernel<0, 0, 0, DD><<<gBig, 256, 0, stream>>>(xb, vwT,  v_b,  nullptr, Vmat,   nullptr, DD);
  gemm_wmma_kernel<0, 0, 0, DD><<<gEnc, 256, 0, stream>>>(hk, kw2T, k_b2, nullptr, kenc,   nullptr, 64);
  gemm_wmma_kernel<0, 0, 0, DD><<<gEnc, 256, 0, stream>>>(hq, qw2T, q_b2, nullptr, qenc,   nullptr, 64);

  // ---- 2: L2 normalize key/query rows ----
  l2norm_kernel<<<MM / 8, 256, 0, stream>>>(kenc, MM);
  l2norm_kernel<<<MM / 8, 256, 0, stream>>>(qenc, MM);

  // ---- 3: chunked phasor scan (128KB complex state in LDS) ----
  size_t shm1 = (size_t)(2 * PP * DD + 2 * PP) * 4;                 // 131328 B
  size_t shm3 = (size_t)(2 * PP * DD + 4 * PP + 32) * 4;            // 131712 B
  chunk_sum_kernel<<<dim3(CC, BB), DD, shm1, stream>>>(kenc, Vmat, cstate);
  chunk_prefix_kernel<<<dim3((2 * PP * DD + 255) / 256, BB), 256, 0, stream>>>(cstate);
  scan_ln_kernel<<<dim3(CC, BB), DD, shm3, stream>>>(kenc, qenc, Vmat, cstate, ln_g, ln_b, rln);

  // ---- 4: output projection + residual (WMMA, f32 epilogue) ----
  gemm_wmma_kernel<0, 0, 1, DD><<<gBig, 256, 0, stream>>>(rln, owT, o_b, x, out, nullptr, DD);
}
